// SCN3d_48352741819130
// MI455X (gfx1250) — matrix operand order repaired and checked by reading, not compiled, not run on hardware
//
#include <hip/hip_runtime.h>
#include <stdint.h>
#include <stddef.h>

// ---------------------------------------------------------------------------
// Sparse 3D U-Net (submanifold) for MI455X / gfx1250.
// Gather-GEMM-scatter over active voxels only; bf16 WMMA 16x16x32, f32 accum.
// v3: branch-free gathers (zero-row base-pointer select instead of exec-masked
//     loads), clamped coord fetch, 4 waves/block (one M-supertile per wave).
// ---------------------------------------------------------------------------

typedef __attribute__((ext_vector_type(16))) __bf16 v16bf;
typedef __attribute__((ext_vector_type(8)))  float  v8f;

#define BN_EPS 0.001f

#define MODE_SAME 0
#define MODE_DOWN 1
#define MODE_UP   2
#define MODE_ONE  3

#define MT 4            // M-tiles (16 rows each) per wave
#define WPB 4           // waves per block

__device__ __forceinline__ unsigned short f2bf(float f) {
    union { float f; uint32_t u; } c; c.f = f;
    uint32_t u = c.u;
    uint32_t r = (u + 0x7FFFu + ((u >> 16) & 1u)) >> 16;  // RNE
    return (unsigned short)r;
}
__device__ __forceinline__ float bf2f(unsigned short s) {
    union { float f; uint32_t u; } c; c.u = ((uint32_t)s) << 16;
    return c.f;
}

// ---------------- utility kernels ----------------
__global__ void fill_int_kernel(int* p, int n, int val) {
    int i = blockIdx.x * blockDim.x + threadIdx.x;
    if (i < n) p[i] = val;
}
__global__ void write_int_kernel(int* p, int val) { *p = val; }

// idx0[lin(coord_i)] = i ; coords0[i] = packed (x<<20|y<<10|z)
__global__ void build_idx0_kernel(const int* __restrict__ coords4, int N, int D,
                                  int* __restrict__ idx0, int* __restrict__ coords0) {
    int i = blockIdx.x * blockDim.x + threadIdx.x;
    if (i >= N) return;
    int x = coords4[i * 4 + 1], y = coords4[i * 4 + 2], z = coords4[i * 4 + 3];
    idx0[(x * D + y) * D + z] = i;
    coords0[i] = (x << 20) | (y << 10) | z;
}

// down_mask: coarse cell active iff any fine voxel in its 3^3 stride-2 window
__global__ void mark_active_kernel(const int* __restrict__ idxf, int Df, int Dc,
                                   int* __restrict__ idxc) {
    int e = blockIdx.x * blockDim.x + threadIdx.x;
    int cells = Dc * Dc * Dc;
    if (e >= cells) return;
    int z = e % Dc, y = (e / Dc) % Dc, x = e / (Dc * Dc);
    int act = 0;
    for (int kd = 0; kd < 3; ++kd)
        for (int kh = 0; kh < 3; ++kh)
            for (int kw = 0; kw < 3; ++kw) {
                int qx = 2 * x + kd - 1, qy = 2 * y + kh - 1, qz = 2 * z + kw - 1;
                if (qx >= 0 && qx < Df && qy >= 0 && qy < Df && qz >= 0 && qz < Df)
                    if (idxf[((size_t)qx * Df + qy) * Df + qz] >= 0) act = 1;
            }
    idxc[e] = act;
}

// Deterministic compaction: flags -> row ids (-1 inactive) + packed coords + count
__global__ void scan_assign_kernel(int* __restrict__ g_idx, int cells, int Dc,
                                   int* __restrict__ coords, int* __restrict__ pcount) {
    __shared__ int sh[1024];
    __shared__ int running;
    int t = threadIdx.x;
    if (t == 0) running = 0;
    __syncthreads();
    for (int base = 0; base < cells; base += 1024) {
        int cell = base + t;
        int v = (cell < cells) ? g_idx[cell] : 0;
        sh[t] = v;
        __syncthreads();
        for (int off = 1; off < 1024; off <<= 1) {
            int val = (t >= off) ? sh[t - off] : 0;
            __syncthreads();
            sh[t] += val;
            __syncthreads();
        }
        int incl = sh[t];
        int total = sh[1023];
        int rb = running;
        if (cell < cells) {
            if (v) {
                int row = rb + incl - v;
                g_idx[cell] = row;
                int z = cell % Dc, y = (cell / Dc) % Dc, x = cell / (Dc * Dc);
                coords[row] = (x << 20) | (y << 10) | z;
            } else {
                g_idx[cell] = -1;
            }
        }
        __syncthreads();
        if (t == 0) running += total;
        __syncthreads();
    }
    if (t == 0) *pcount = running;
}

// Repack OIDHW f32 weights into WMMA B-operand lane layout, bf16.
// Tile (tap,kc,cot): 512 halves; lane l (=(e>>4)&31), j (=e&15):
//   N = l&15, K = (l>>4)*16 + j  -> B[K][N] = W[o=cot*16+N][i=kc*32+K]
__global__ void repack_w_kernel(const float* __restrict__ W, unsigned short* __restrict__ out,
                                int Cin, int Cout, int taps) {
    int KC = (Cin + 31) >> 5, COT = Cout >> 4;
    int total = taps * KC * COT * 512;
    int e = blockIdx.x * blockDim.x + threadIdx.x;
    if (e >= total) return;
    int j = e & 15;
    int lane = (e >> 4) & 31;
    int tile = e >> 9;
    int cot = tile % COT;
    int kc = (tile / COT) % KC;
    int tap = tile / (COT * KC);
    int Ncol = lane & 15;
    int K = ((lane >> 4) << 4) + j;
    int o = cot * 16 + Ncol;
    int i = kc * 32 + K;
    float v = (i < Cin) ? W[((size_t)o * Cin + i) * taps + tap] : 0.f;
    out[e] = f2bf(v);
}

// Stem GEMM: f[i,c] = sum_k feat[i,k] * Wstem[k,c] + b[c]   (tiny, memory-bound)
__global__ void stem_kernel(const float* __restrict__ F, const float* __restrict__ W,
                            const float* __restrict__ b, float* __restrict__ out, int N) {
    int e = blockIdx.x * blockDim.x + threadIdx.x;
    if (e >= N * 16) return;
    int c = e & 15, r = e >> 4;
    float s = b[c];
#pragma unroll
    for (int k = 0; k < 16; ++k) s += F[r * 16 + k] * W[k * 16 + c];
    out[e] = s;
}

// ---------------- BN (deterministic two-stage) ----------------
__global__ void stats_partial_kernel(const float* __restrict__ x, const int* __restrict__ pM,
                                     int C, int cshift, int rpb, float* __restrict__ part) {
    __shared__ float sh[512];
    int M = *pM;
    int t = threadIdx.x;
    int G = 256 >> cshift;
    int c = t & (C - 1), g = t >> cshift;
    float s = 0.f, q = 0.f;
    int base = blockIdx.x * rpb;
    for (int i = g; i < rpb; i += G) {
        int r = base + i;
        if (r < M) { float v = x[((size_t)r << cshift) + c]; s += v; q += v * v; }
    }
    sh[t] = s; sh[256 + t] = q;
    __syncthreads();
    for (int st = 128; st >= C; st >>= 1) {
        if (t < st) { sh[t] += sh[t + st]; sh[256 + t] += sh[256 + t + st]; }
        __syncthreads();
    }
    if (t < C) {
        part[(size_t)blockIdx.x * 2 * C + t] = sh[t];
        part[(size_t)blockIdx.x * 2 * C + C + t] = sh[256 + t];
    }
}

__global__ void stats_final_kernel(const float* __restrict__ part, int nblk, int C,
                                   const int* __restrict__ pM, float* __restrict__ stats) {
    int t = threadIdx.x;
    int m = *pM; if (m < 1) m = 1;
    float n = (float)m;
    for (int c = t; c < C; c += blockDim.x) {
        float s = 0.f, q = 0.f;
        for (int b = 0; b < nblk; ++b) {              // fixed order -> deterministic
            s += part[(size_t)b * 2 * C + c];
            q += part[(size_t)b * 2 * C + C + c];
        }
        float mean = s / n;
        float var = q / n - mean * mean;
        if (var < 0.f) var = 0.f;
        stats[c] = mean;
        stats[128 + c] = rsqrtf(var + BN_EPS);
    }
}

// mode 0: out_bf = relu(norm)            (bn_relu)
// mode 1: out_bf = add + relu(norm)      (t = t + bn_relu(...))
// mode 2: out_f32 = relu(norm + add)     (final: relu(masked_bn(y) + t))
__global__ void bn_apply_kernel(const float* __restrict__ x, const unsigned short* __restrict__ addb,
                                unsigned short* __restrict__ outb, float* __restrict__ outf,
                                const float* __restrict__ stats, const int* __restrict__ pM,
                                int C, int cshift, int mode) {
    int M = *pM;
    int e = blockIdx.x * blockDim.x + threadIdx.x;
    int r = e >> cshift;
    if (r >= M) return;
    int c = e & (C - 1);
    float v = (x[e] - stats[c]) * stats[128 + c];
    if (mode == 0) {
        v = fmaxf(v, 0.f);
        outb[e] = f2bf(v);
    } else if (mode == 1) {
        v = fmaxf(v, 0.f) + bf2f(addb[e]);
        outb[e] = f2bf(v);
    } else {
        v = fmaxf(v + bf2f(addb[e]), 0.f);
        outf[e] = v;
    }
}

// ---------------- implicit-GEMM conv via bf16 WMMA ----------------
union ABTile { v16bf v; uint4 q[2]; unsigned short s[16]; };

// 4 waves/block; each wave computes MT x (16 rows) x 16 cout columns.
// B tile fetched once per (tap, kc) and fed to MT independent WMMAs.
// Dead gathers read a shared zeroed row via base-pointer select (no EXEC churn).
__global__ __launch_bounds__(32 * WPB)
void conv_wmma_kernel(const unsigned short* __restrict__ A,   // bf16 [Min x Cin]
                      const unsigned short* __restrict__ Wr,  // repacked bf16
                      const unsigned short* __restrict__ zrow,// 256B of zeros
                      float* __restrict__ Out,                // f32  [Mout x Cout]
                      const int* __restrict__ coords_out, const int* __restrict__ pMout,
                      const int* __restrict__ idx_in,
                      int Cin, int Cout, int Din, int Dout, int mode) {
    int Mout = *pMout;
    int wave = threadIdx.x >> 5;
    int tile = blockIdx.x * WPB + wave;
    if (tile * (16 * MT) >= Mout) return;   // wave-uniform exit; EXEC full
    int cot = blockIdx.y;
    int l = threadIdx.x & 31;
    int Mrow = l & 15;
    int KC = (Cin + 31) >> 5;
    int COT = Cout >> 4;
    int bK = (l >> 4) << 4;                 // A-operand K base per half-wave: 0 or 16

    int rrow[MT], px[MT], py[MT], pz[MT];
#pragma unroll
    for (int mt = 0; mt < MT; ++mt) {
        int r = tile * (16 * MT) + mt * 16 + Mrow;
        rrow[mt] = r;
        int rc = (r < Mout) ? r : (Mout - 1);       // clamp: no exec-masked load
        int lp = coords_out[rc];
        px[mt] = (lp >> 20) & 1023;
        py[mt] = (lp >> 10) & 1023;
        pz[mt] = lp & 1023;
    }

    v8f acc[MT];
#pragma unroll
    for (int mt = 0; mt < MT; ++mt) acc[mt] = (v8f){0.f,0.f,0.f,0.f,0.f,0.f,0.f,0.f};

    int ntap = (mode == MODE_ONE) ? 1 : 27;

    for (int tap = 0; tap < ntap; ++tap) {
        int kd = tap / 9, kh = (tap / 3) % 3, kw = tap % 3;   // uniform scalars
        int nr[MT];
        int alive = 0;
#pragma unroll
        for (int mt = 0; mt < MT; ++mt) {
            nr[mt] = -1;
            if (mode == MODE_ONE) {
                nr[mt] = (rrow[mt] < Mout) ? rrow[mt] : (Mout - 1);
            } else {
                int qx, qy, qz;
                bool ok;
                if (mode == MODE_SAME) {
                    qx = px[mt] + kd - 1; qy = py[mt] + kh - 1; qz = pz[mt] + kw - 1;
                    ok = (qx >= 0) & (qx < Din) & (qy >= 0) & (qy < Din) &
                         (qz >= 0) & (qz < Din);
                } else if (mode == MODE_DOWN) {
                    qx = 2 * px[mt] + kd - 1; qy = 2 * py[mt] + kh - 1;
                    qz = 2 * pz[mt] + kw - 1;
                    ok = (qx >= 0) & (qx < Din) & (qy >= 0) & (qy < Din) &
                         (qz >= 0) & (qz < Din);
                } else { // MODE_UP: transposed conv, lhs_dilation=2, pad (1,2)
                    int jx = px[mt] - 1 + kd, jy = py[mt] - 1 + kh, jz = pz[mt] - 1 + kw;
                    ok = (jx >= 0) & (jx <= 2 * Din - 2) & ((jx & 1) == 0) &
                         (jy >= 0) & (jy <= 2 * Din - 2) & ((jy & 1) == 0) &
                         (jz >= 0) & (jz <= 2 * Din - 2) & ((jz & 1) == 0);
                    qx = jx >> 1; qy = jy >> 1; qz = jz >> 1;
                }
                if (ok) nr[mt] = idx_in[((size_t)qx * Din + qy) * Din + qz];
            }
            alive |= (nr[mt] >= 0);
        }
        // Uniform skip of fully-dead taps (keeps EXEC all-ones for WMMA).
        if (__ballot(alive) == 0ull) continue;

        // Base-pointer select: dead rows gather from the shared zero block.
        const unsigned short* rowp[MT];
#pragma unroll
        for (int mt = 0; mt < MT; ++mt)
            rowp[mt] = (nr[mt] >= 0) ? (A + (size_t)nr[mt] * Cin) : zrow;

        const unsigned short* wbase =
            Wr + (((size_t)tap * KC) * COT + cot) * 512 + ((size_t)l << 4);

        for (int kc = 0; kc < KC; ++kc) {
            ABTile b;
            const unsigned short* wp = wbase + ((size_t)kc * COT << 9);
            b.q[0] = *(const uint4*)(wp);
            b.q[1] = *(const uint4*)(wp + 8);
            int k0 = (kc << 5) + bK;  // A layout: lane l holds K {k0..k0+15}
            int k1 = k0 + 8;
            bool v0 = (k0 < Cin), v1 = (k1 < Cin);        // uniform per half-wave
            uint4 z4 = {0u, 0u, 0u, 0u};
#pragma unroll
            for (int mt = 0; mt < MT; ++mt) {
                ABTile a;
                // dead lanes read zrow (offset <= 240B, inside the 256B block)
                a.q[0] = v0 ? *(const uint4*)(rowp[mt] + (nr[mt] >= 0 ? k0 : 0)) : z4;
                a.q[1] = v1 ? *(const uint4*)(rowp[mt] + (nr[mt] >= 0 ? k1 : 16)) : z4;
#if defined(__gfx1250__)
                acc[mt] = __builtin_amdgcn_wmma_f32_16x16x32_bf16(
                    false, a.v, false, b.v, (short)0, acc[mt], false, false);
#else
                acc[mt][0] += (float)a.s[0] * (float)b.s[0];  // host-pass placeholder
#endif
            }
        }
    }
    // D layout: VGPR e -> M = e + (lane<16 ? 0 : 8); N = lane&15
    int obase = (l >> 4) << 3;
    int ncol = l & 15;
#pragma unroll
    for (int mt = 0; mt < MT; ++mt) {
#pragma unroll
        for (int e2 = 0; e2 < 8; ++e2) {
            int rr = tile * (16 * MT) + mt * 16 + obase + e2;
            if (rr < Mout) Out[(size_t)rr * Cout + cot * 16 + ncol] = acc[mt][e2];
        }
    }
}

// ---------------------------------------------------------------------------
extern "C" void kernel_launch(void* const* d_in, const int* in_sizes, int n_in,
                              void* d_out, int out_size, void* d_ws, size_t ws_size,
                              hipStream_t stream) {
    const int N  = in_sizes[1] / 4;     // voxel_coords is N x 4
    const int D0 = 96, D1 = 48, D2 = 24, D3 = 12;   // voxel_dim per setup_inputs
    const int CAP1 = D1 * D1 * D1, CAP2 = D2 * D2 * D2, CAP3 = D3 * D3 * D3;

    const float* feat   = (const float*)d_in[0];
    const int*   vcoord = (const int*)d_in[1];
    const float* Wstem  = (const float*)d_in[3];
    const float* bstem  = (const float*)d_in[4];
    const float* Wd0 = (const float*)d_in[5],  *Wd1 = (const float*)d_in[6];
    const float* Wd2 = (const float*)d_in[7];
    const float* Wl0 = (const float*)d_in[8],  *Wl1 = (const float*)d_in[9];
    const float* Wl2 = (const float*)d_in[10], *Wl3 = (const float*)d_in[11];
    const float* Wi1 = (const float*)d_in[12], *Wi2 = (const float*)d_in[13];
    const float* Wi3 = (const float*)d_in[14];
    const float* Wu0 = (const float*)d_in[15], *Wb1 = (const float*)d_in[16];
    const float* Wb2 = (const float*)d_in[17];

    // ---- workspace carve-up (256B aligned) ----
    char* ws = (char*)d_ws;
    size_t off = 0;
    auto alloc = [&](size_t bytes) -> size_t {
        size_t o = off; off = (off + bytes + 255) & ~(size_t)255; return o;
    };
    size_t o_idx0 = alloc((size_t)D0 * D0 * D0 * 4);
    size_t o_idx1 = alloc((size_t)CAP1 * 4);
    size_t o_idx2 = alloc((size_t)CAP2 * 4);
    size_t o_idx3 = alloc((size_t)CAP3 * 4);
    size_t o_crd0 = alloc((size_t)N * 4);
    size_t o_crd1 = alloc((size_t)CAP1 * 4);
    size_t o_crd2 = alloc((size_t)CAP2 * 4);
    size_t o_crd3 = alloc((size_t)CAP3 * 4);
    size_t o_cnt  = alloc(4 * 4);
    size_t o_stat = alloc(256 * 4);
    size_t o_zero = alloc(256);                          // shared zero row (bf16)
    size_t o_part = alloc((size_t)2 * 1024 * 1024);
    size_t o_tmpf = alloc((size_t)CAP1 * 64 * 4);        // f32 conv accum (max)
    size_t o_f0 = alloc((size_t)N * 16 * 2);
    size_t o_x1 = alloc((size_t)CAP1 * 32 * 2);
    size_t o_x2 = alloc((size_t)CAP2 * 64 * 2);
    size_t o_x3 = alloc((size_t)CAP3 * 128 * 2);
    size_t o_t3 = alloc((size_t)CAP3 * 64 * 2);
    size_t o_t2 = alloc((size_t)CAP2 * 64 * 2);
    size_t o_t1 = alloc((size_t)CAP1 * 64 * 2);
    size_t o_t0 = alloc((size_t)N * 64 * 2);
    size_t o_tA = alloc((size_t)N * 64 * 2);
    size_t o_y1 = alloc((size_t)N * 64 * 2);
    auto rsize = [](int Cin, int Cout, int taps) -> size_t {
        return (size_t)taps * ((Cin + 31) / 32) * (Cout / 16) * 512 * 2;
    };
    size_t o_Rd0 = alloc(rsize(16, 32, 27)),  o_Rd1 = alloc(rsize(32, 64, 27));
    size_t o_Rd2 = alloc(rsize(64, 128, 27));
    size_t o_Rl0 = alloc(rsize(16, 64, 1)),   o_Rl1 = alloc(rsize(32, 64, 1));
    size_t o_Rl2 = alloc(rsize(64, 64, 1)),   o_Rl3 = alloc(rsize(128, 64, 1));
    size_t o_Ri1 = alloc(rsize(64, 64, 27)),  o_Ri2 = alloc(rsize(64, 64, 27));
    size_t o_Ri3 = alloc(rsize(64, 64, 27));
    size_t o_Ru0 = alloc(rsize(64, 64, 27)),  o_Rb1 = alloc(rsize(64, 64, 27));
    size_t o_Rb2 = alloc(rsize(64, 64, 27));
    (void)ws_size;

    int* idx0 = (int*)(ws + o_idx0); int* idx1 = (int*)(ws + o_idx1);
    int* idx2 = (int*)(ws + o_idx2); int* idx3 = (int*)(ws + o_idx3);
    int* crd0 = (int*)(ws + o_crd0); int* crd1 = (int*)(ws + o_crd1);
    int* crd2 = (int*)(ws + o_crd2); int* crd3 = (int*)(ws + o_crd3);
    int* cnt  = (int*)(ws + o_cnt);
    float* stats = (float*)(ws + o_stat);
    float* part  = (float*)(ws + o_part);
    float* tmpf  = (float*)(ws + o_tmpf);
    unsigned short* zrow = (unsigned short*)(ws + o_zero);
    unsigned short* f0 = (unsigned short*)(ws + o_f0);
    unsigned short* x1 = (unsigned short*)(ws + o_x1);
    unsigned short* x2 = (unsigned short*)(ws + o_x2);
    unsigned short* x3 = (unsigned short*)(ws + o_x3);
    unsigned short* t3 = (unsigned short*)(ws + o_t3);
    unsigned short* t2 = (unsigned short*)(ws + o_t2);
    unsigned short* t1 = (unsigned short*)(ws + o_t1);
    unsigned short* t0 = (unsigned short*)(ws + o_t0);
    unsigned short* tA = (unsigned short*)(ws + o_tA);
    unsigned short* y1 = (unsigned short*)(ws + o_y1);

    auto cdiv = [](int a, int b) { return (a + b - 1) / b; };
    auto log2i = [](int c) { int s = 0; while ((1 << s) < c) ++s; return s; };

    // ---- index volumes / active sets / zero row ----
    fill_int_kernel<<<cdiv(D0 * D0 * D0, 256), 256, 0, stream>>>(idx0, D0 * D0 * D0, -1);
    fill_int_kernel<<<1, 64, 0, stream>>>((int*)zrow, 64, 0);
    write_int_kernel<<<1, 1, 0, stream>>>(cnt + 0, N);
    build_idx0_kernel<<<cdiv(N, 256), 256, 0, stream>>>(vcoord, N, D0, idx0, crd0);
    mark_active_kernel<<<cdiv(CAP1, 256), 256, 0, stream>>>(idx0, D0, D1, idx1);
    scan_assign_kernel<<<1, 1024, 0, stream>>>(idx1, CAP1, D1, crd1, cnt + 1);
    mark_active_kernel<<<cdiv(CAP2, 256), 256, 0, stream>>>(idx1, D1, D2, idx2);
    scan_assign_kernel<<<1, 1024, 0, stream>>>(idx2, CAP2, D2, crd2, cnt + 2);
    mark_active_kernel<<<cdiv(CAP3, 256), 256, 0, stream>>>(idx2, D2, D3, idx3);
    scan_assign_kernel<<<1, 1024, 0, stream>>>(idx3, CAP3, D3, crd3, cnt + 3);

    // ---- weight repack (B-operand lane layout, bf16) ----
    auto repack = [&](const float* W, size_t oR, int Cin, int Cout, int taps) {
        int total = taps * ((Cin + 31) / 32) * (Cout / 16) * 512;
        repack_w_kernel<<<cdiv(total, 256), 256, 0, stream>>>(
            W, (unsigned short*)(ws + oR), Cin, Cout, taps);
    };
    repack(Wd0, o_Rd0, 16, 32, 27);  repack(Wd1, o_Rd1, 32, 64, 27);
    repack(Wd2, o_Rd2, 64, 128, 27);
    repack(Wl0, o_Rl0, 16, 64, 1);   repack(Wl1, o_Rl1, 32, 64, 1);
    repack(Wl2, o_Rl2, 64, 64, 1);   repack(Wl3, o_Rl3, 128, 64, 1);
    repack(Wi1, o_Ri1, 64, 64, 27);  repack(Wi2, o_Ri2, 64, 64, 27);
    repack(Wi3, o_Ri3, 64, 64, 27);
    repack(Wu0, o_Ru0, 64, 64, 27);  repack(Wb1, o_Rb1, 64, 64, 27);
    repack(Wb2, o_Rb2, 64, 64, 27);

    // ---- fused BN helper (deterministic two-stage stats + apply) ----
    auto run_bn = [&](const float* x, const unsigned short* addb, unsigned short* outb,
                      float* outf, const int* pM, int cap, int C, int mode) {
        int cs = log2i(C);
        int G = 256 >> cs, rpb = G * 16, nblk = cdiv(cap, rpb);
        stats_partial_kernel<<<nblk, 256, 0, stream>>>(x, pM, C, cs, rpb, part);
        stats_final_kernel<<<1, 128, 0, stream>>>(part, nblk, C, pM, stats);
        bn_apply_kernel<<<cdiv(cap * C, 256), 256, 0, stream>>>(
            x, addb, outb, outf, stats, pM, C, cs, mode);
    };
    auto run_conv = [&](const unsigned short* A, size_t oR, const int* coords,
                        const int* pM, int cap, const int* idxin,
                        int Cin, int Cout, int Din, int Dout, int mode) {
        dim3 g(cdiv(cap, 16 * MT * WPB), Cout / 16);
        conv_wmma_kernel<<<g, 32 * WPB, 0, stream>>>(
            A, (const unsigned short*)(ws + oR), zrow, tmpf, coords, pM, idxin,
            Cin, Cout, Din, Dout, mode);
    };

    // ---- stem: GEMM + BN + relu -> f0 (bf16, rows == input voxel order) ----
    stem_kernel<<<cdiv(N * 16, 256), 256, 0, stream>>>(feat, Wstem, bstem, tmpf, N);
    run_bn(tmpf, nullptr, f0, nullptr, cnt + 0, N, 16, 0);

    // ---- encoder ----
    run_conv(f0, o_Rd0, crd1, cnt + 1, CAP1, idx0, 16, 32, D0, D1, MODE_DOWN);
    run_bn(tmpf, nullptr, x1, nullptr, cnt + 1, CAP1, 32, 0);
    run_conv(x1, o_Rd1, crd2, cnt + 2, CAP2, idx1, 32, 64, D1, D2, MODE_DOWN);
    run_bn(tmpf, nullptr, x2, nullptr, cnt + 2, CAP2, 64, 0);
    run_conv(x2, o_Rd2, crd3, cnt + 3, CAP3, idx2, 64, 128, D2, D3, MODE_DOWN);
    run_bn(tmpf, nullptr, x3, nullptr, cnt + 3, CAP3, 128, 0);

    // ---- FPN decoder ----
    run_conv(x3, o_Rl3, crd3, cnt + 3, CAP3, idx3, 128, 64, D3, D3, MODE_ONE);
    run_bn(tmpf, nullptr, t3, nullptr, cnt + 3, CAP3, 64, 0);

    run_conv(t3, o_Ri3, crd2, cnt + 2, CAP2, idx3, 64, 64, D3, D2, MODE_UP);
    run_bn(tmpf, nullptr, t2, nullptr, cnt + 2, CAP2, 64, 0);
    run_conv(x2, o_Rl2, crd2, cnt + 2, CAP2, idx2, 64, 64, D2, D2, MODE_ONE);
    run_bn(tmpf, t2, t2, nullptr, cnt + 2, CAP2, 64, 1);

    run_conv(t2, o_Ri2, crd1, cnt + 1, CAP1, idx2, 64, 64, D2, D1, MODE_UP);
    run_bn(tmpf, nullptr, t1, nullptr, cnt + 1, CAP1, 64, 0);
    run_conv(x1, o_Rl1, crd1, cnt + 1, CAP1, idx1, 32, 64, D1, D1, MODE_ONE);
    run_bn(tmpf, t1, t1, nullptr, cnt + 1, CAP1, 64, 1);

    run_conv(t1, o_Ri1, crd0, cnt + 0, N, idx1, 64, 64, D1, D0, MODE_UP);
    run_bn(tmpf, nullptr, t0, nullptr, cnt + 0, N, 64, 0);
    run_conv(f0, o_Rl0, crd0, cnt + 0, N, idx0, 16, 64, D0, D0, MODE_ONE);
    run_bn(tmpf, t0, t0, nullptr, cnt + 0, N, 64, 1);

    // ---- head ----
    run_conv(t0, o_Ru0, crd0, cnt + 0, N, idx0, 64, 64, D0, D0, MODE_SAME);
    run_bn(tmpf, nullptr, tA, nullptr, cnt + 0, N, 64, 0);
    run_conv(tA, o_Rb1, crd0, cnt + 0, N, idx0, 64, 64, D0, D0, MODE_SAME);
    run_bn(tmpf, nullptr, y1, nullptr, cnt + 0, N, 64, 0);
    run_conv(y1, o_Rb2, crd0, cnt + 0, N, idx0, 64, 64, D0, D0, MODE_SAME);
    run_bn(tmpf, tA, nullptr, (float*)d_out, cnt + 0, N, 64, 2);  // relu(bn(y)+t)
    (void)out_size; (void)n_in;
}
